// BiMamba_59631325938215
// MI455X (gfx1250) — compile-verified, matrix operand order checked by reading, not run on hardware
//
#include <hip/hip_runtime.h>
#include <hip/hip_bf16.h>
#include <math.h>

#define BB 4
#define TT 2048
#define DF 352          // D_FULL
#define DI 704          // D_INNER
#define DSTATE 16
#define DTRANK 22
#define KCONV 4
#define NDEPTH 5
#define MROWS (BB*TT)   // 8192
#define XDBL_LD 64      // padded 54 -> 64: [0:22)=dt_in, [32:48)=B, [48:64)=C
#define BT_STRIDE 40    // padded K-stride of transposed B tile (16B aligned rows)

typedef __attribute__((ext_vector_type(16))) _Float16 v16h;
typedef __attribute__((ext_vector_type(8)))  _Float16 h8;
typedef __attribute__((ext_vector_type(4)))  _Float16 h4;
typedef __attribute__((ext_vector_type(8)))  float    v8f;

// ---------------------------------------------------------------------------
// WMMA GEMM: C[M x N] = act(A[M x K](f32) @ W[K x N](f16) + bias) + resid
// Block = 128 threads (4 waves). Block tile 128(M) x 32(N), K step 32.
// Each wave owns a 32x32 patch: 2 A-frags x 2 B-frags = 4 WMMA / K-step.
// Requires M%128==0, N%32==0, K%32==0 (guaranteed by padding).
// act: 0=none, 1=GELU(exact erf), 2=softplus
// ---------------------------------------------------------------------------
__global__ __launch_bounds__(128)
void gemm_wmma(const float* __restrict__ A, int lda,
               const _Float16* __restrict__ W, int ldb,
               const float* __restrict__ bias,
               const float* resid,
               float* C, int ldc,
               int K, int act)
{
    __shared__ _Float16 As[128 * 32];           // [row][k], k contiguous (8KB)
    __shared__ _Float16 Bt[32 * BT_STRIDE];     // transposed: [col][k] (2.5KB)
    const int tid  = threadIdx.x;
    const int wave = tid >> 5;
    const int lane = tid & 31;
    const int r    = lane & 15;
    const int hi   = lane >> 4;                 // half-wave select (ISA frag layout)
    const int tileM = blockIdx.y * 128;
    const int tileN = blockIdx.x * 32;

    v8f acc[2][2] = {{{}, {}}, {{}, {}}};

    for (int k0 = 0; k0 < K; k0 += 32) {
        // ---- stage A tile 128x32: issue all 8 b128 loads, then convert+store
        float4 va[8];
        #pragma unroll
        for (int it = 0; it < 8; ++it) {
            int i = tid * 4 + it * 512;
            int rr = i >> 5, cc = i & 31;
            va[it] = *(const float4*)(A + (size_t)(tileM + rr) * lda + (k0 + cc));
        }
        // ---- B tile 32x32: one b128 load per thread (fp16 weights)
        h8 vb;
        {
            int i = tid * 8;                    // 1024 halves / 128 thr = 8 each
            int rr = i >> 5, cc = i & 31;
            vb = *(const h8*)(W + (size_t)(k0 + rr) * ldb + (tileN + cc));
        }
        #pragma unroll
        for (int it = 0; it < 8; ++it) {
            int i = tid * 4 + it * 512;
            h4 p; p[0] = (_Float16)va[it].x; p[1] = (_Float16)va[it].y;
                  p[2] = (_Float16)va[it].z; p[3] = (_Float16)va[it].w;
            *(h4*)&As[i] = p;
        }
        {
            int i = tid * 8;
            int rr = i >> 5, cc = i & 31;
            #pragma unroll
            for (int q = 0; q < 8; ++q)
                Bt[(cc + q) * BT_STRIDE + rr] = vb[q];
        }
        __syncthreads();

        // prefetch next A tile rows (global_prefetch_b8)
        if (k0 + 32 < K)
            __builtin_prefetch(&A[(size_t)(tileM + tid) * lda + (k0 + 32)], 0, 1);

        // ---- B fragments (shared across both M sub-tiles)
        v16h bf[2];
        #pragma unroll
        for (int n = 0; n < 2; ++n) {
            const int col = 16 * n + r;
            h8 blo = *(const h8*)&Bt[col * BT_STRIDE + hi * 8];
            h8 bhi = *(const h8*)&Bt[col * BT_STRIDE + 16 + hi * 8];
            bf[n] = __builtin_shufflevector(blo, bhi,
                        0, 1, 2, 3, 4, 5, 6, 7, 8, 9, 10, 11, 12, 13, 14, 15);
        }
        // ---- A fragments + 4 WMMA
        #pragma unroll
        for (int m = 0; m < 2; ++m) {
            const int arow = 32 * wave + 16 * m + r;
            h8 alo = *(const h8*)&As[arow * 32 + hi * 8];
            h8 ahi = *(const h8*)&As[arow * 32 + 16 + hi * 8];
            v16h af = __builtin_shufflevector(alo, ahi,
                        0, 1, 2, 3, 4, 5, 6, 7, 8, 9, 10, 11, 12, 13, 14, 15);
            #pragma unroll
            for (int n = 0; n < 2; ++n)
                acc[m][n] = __builtin_amdgcn_wmma_f32_16x16x32_f16(
                    false, af, false, bf[n], (short)0, acc[m][n], false, false);
        }
        __syncthreads();
    }

    // ---- epilogue: D VGPR j, lane l -> row = 8*hi + j, col = r (ISA C/D layout)
    #pragma unroll
    for (int m = 0; m < 2; ++m) {
        #pragma unroll
        for (int n = 0; n < 2; ++n) {
            const int col = tileN + 16 * n + r;
            #pragma unroll
            for (int j = 0; j < 8; ++j) {
                int row = tileM + 32 * wave + 16 * m + hi * 8 + j;
                float v = acc[m][n][j];
                if (bias) v += bias[col];
                if (act == 1) {                       // exact GELU
                    v = 0.5f * v * (1.0f + erff(v * 0.70710678f));
                } else if (act == 2) {                // softplus
                    v = (v > 20.0f) ? v : log1pf(__expf(v));
                }
                if (resid) v += resid[(size_t)row * ldc + col];
                C[(size_t)row * ldc + col] = v;
            }
        }
    }
}

// ---------------------------------------------------------------------------
// fp32 -> fp16 weight convert (once per layer per GEMM, L2-resident)
// ---------------------------------------------------------------------------
__global__ void cvt_w_kernel(const float* __restrict__ W, _Float16* __restrict__ Wh, int n)
{
    int i = blockIdx.x * blockDim.x + threadIdx.x;
    if (i < n) Wh[i] = (_Float16)W[i];
}

// ---------------------------------------------------------------------------
// Embedding: concat(poi_emb[token], time_emb[ctx0], [ctx1,ctx2]@space_W + b)
// ---------------------------------------------------------------------------
__global__ void embed_kernel(float* __restrict__ out,
                             const int* __restrict__ x,
                             const int* __restrict__ ctx,
                             const float* __restrict__ poi,
                             const float* __restrict__ tim,
                             const float* __restrict__ sW,
                             const float* __restrict__ sb,
                             int reverse)
{
    int idx = blockIdx.x * blockDim.x + threadIdx.x;
    if (idx >= MROWS * DF) return;
    int m = idx / DF, j = idx % DF;
    int b = m / TT, t = m % TT;
    int ts = reverse ? (TT - 1 - t) : t;
    int src = b * TT + ts;
    float v;
    if (j < 256) {
        v = poi[(size_t)x[src] * 256 + j];
    } else if (j < 288) {
        v = tim[(size_t)ctx[src] * 32 + (j - 256)];
    } else {
        int jj = j - 288;
        float c1 = (float)ctx[(size_t)BB * TT + src];
        float c2 = (float)ctx[(size_t)2 * BB * TT + src];
        v = c1 * sW[jj] + c2 * sW[64 + jj] + sb[jj];
    }
    out[(size_t)m * DF + j] = v;
}

// ---------------------------------------------------------------------------
// LayerNorm over last dim (DF), one block per row
// ---------------------------------------------------------------------------
__global__ __launch_bounds__(128)
void layernorm_kernel(const float* __restrict__ in, float* __restrict__ out,
                      const float* __restrict__ g, const float* __restrict__ bta)
{
    __shared__ float s1[128], s2[128];
    int row = blockIdx.x;
    int tid = threadIdx.x;
    const float* xr = in + (size_t)row * DF;
    float a = 0.f, b = 0.f;
    for (int j = tid; j < DF; j += 128) { float v = xr[j]; a += v; b += v * v; }
    s1[tid] = a; s2[tid] = b;
    __syncthreads();
    for (int s = 64; s > 0; s >>= 1) {
        if (tid < s) { s1[tid] += s1[tid + s]; s2[tid] += s2[tid + s]; }
        __syncthreads();
    }
    float mean = s1[0] / DF;
    float var  = s2[0] / DF - mean * mean;
    float rstd = rsqrtf(var + 1e-5f);
    for (int j = tid; j < DF; j += 128)
        out[(size_t)row * DF + j] = (xr[j] - mean) * rstd * g[j] + bta[j];
}

// ---------------------------------------------------------------------------
// Depthwise causal conv (K=4) over xh half of xz, + bias, SiLU
// ---------------------------------------------------------------------------
__global__ void conv_silu_kernel(const float* __restrict__ xz,  // M x 2*DI
                                 const float* __restrict__ cw,  // DI x 4
                                 const float* __restrict__ cb,  // DI
                                 float* __restrict__ out)       // M x DI
{
    int idx = blockIdx.x * blockDim.x + threadIdx.x;
    if (idx >= MROWS * DI) return;
    int m = idx / DI, d = idx % DI;
    int t = m % TT;
    float acc = cb[d];
    #pragma unroll
    for (int k = 0; k < KCONV; ++k) {
        int tt = t + k - (KCONV - 1);
        if (tt >= 0)
            acc += xz[(size_t)(m + k - (KCONV - 1)) * (2 * DI) + d] * cw[d * KCONV + k];
    }
    acc = acc / (1.0f + __expf(-acc));  // SiLU
    out[(size_t)m * DI + d] = acc;
}

// ---------------------------------------------------------------------------
// Selective scan: one thread per (b, d) channel; h[16] in registers.
// B at xdbl[32..47], C at xdbl[48..63] -> 16B-aligned float4 loads.
// y = (scan output + xh*D) * silu(z)
// ---------------------------------------------------------------------------
__global__ void scan_kernel(const float* __restrict__ dtb,    // M x DI (softplus'd)
                            const float* __restrict__ xh,     // M x DI
                            const float* __restrict__ xdbl,   // M x 64
                            const float* __restrict__ xz,     // M x 2*DI (z at DI+d)
                            const float* __restrict__ A_log,  // DI x 16
                            const float* __restrict__ Dp,     // DI
                            float* __restrict__ y)            // M x DI
{
    int idx = blockIdx.x * blockDim.x + threadIdx.x;
    if (idx >= BB * DI) return;
    int b = idx / DI, d = idx % DI;
    float A[DSTATE], h[DSTATE];
    #pragma unroll
    for (int s = 0; s < DSTATE; ++s) {
        A[s] = -__expf(A_log[(size_t)d * DSTATE + s]);
        h[s] = 0.f;
    }
    float Dd = Dp[d];
    for (int t = 0; t < TT; ++t) {
        size_t m = (size_t)b * TT + t;
        float dt = dtb[m * DI + d];
        float xv = xh[m * DI + d];
        float zv = xz[m * (2 * DI) + DI + d];
        const float4* Bp = (const float4*)(xdbl + m * XDBL_LD + 32);
        const float4* Cp = (const float4*)(xdbl + m * XDBL_LD + 48);
        float Bv[DSTATE], Cv[DSTATE];
        #pragma unroll
        for (int q = 0; q < 4; ++q) {
            *(float4*)&Bv[q * 4] = Bp[q];
            *(float4*)&Cv[q * 4] = Cp[q];
        }
        float acc = 0.f;
        #pragma unroll
        for (int s = 0; s < DSTATE; ++s) {
            float dA = __expf(dt * A[s]);
            h[s] = dA * h[s] + dt * Bv[s] * xv;
            acc += h[s] * Cv[s];
        }
        acc += xv * Dd;
        acc *= zv / (1.0f + __expf(-zv));
        y[m * DI + d] = acc;
    }
}

// ---------------------------------------------------------------------------
// Pad + fp16-convert x_W (DIx54 -> DIx64, B->cols 32..47, C->48..63) and
// dt_W (22xDI -> 32xDI, zero rows 22..31)
// ---------------------------------------------------------------------------
__global__ void pack_w_kernel(const float* __restrict__ xW,
                              const float* __restrict__ dtW,
                              _Float16* __restrict__ xWp,
                              _Float16* __restrict__ dtWp)
{
    int idx = blockIdx.x * blockDim.x + threadIdx.x;
    const int NXW = DI * 64;
    const int NDW = 32 * DI;
    if (idx < NXW) {
        int rr = idx / 64, cc = idx % 64;
        float v = 0.f;
        if (cc < DTRANK)       v = xW[(size_t)rr * 54 + cc];        // dt_in
        else if (cc >= 32)     v = xW[(size_t)rr * 54 + (cc - 10)]; // B (22..37) / C (38..53)
        xWp[idx] = (_Float16)v;
    } else if (idx < NXW + NDW) {
        int i = idx - NXW;
        int rr = i / DI, cc = i % DI;
        dtWp[i] = (_Float16)((rr < DTRANK) ? dtW[(size_t)rr * DI + cc] : 0.f);
    }
}

// ---------------------------------------------------------------------------
// sum = resid (mode 0)  /  sum += time-reversed resid (mode 1)
// ---------------------------------------------------------------------------
__global__ void accum_kernel(const float* __restrict__ resid, float* __restrict__ sum, int mode)
{
    int idx = blockIdx.x * blockDim.x + threadIdx.x;
    if (idx >= MROWS * DF) return;
    if (mode == 0) {
        sum[idx] = resid[idx];
    } else {
        int m = idx / DF, j = idx % DF;
        int b = m / TT, t = m % TT;
        sum[idx] += resid[((size_t)b * TT + (TT - 1 - t)) * DF + j];
    }
}

// ---------------------------------------------------------------------------
// mean over T -> out (B x DF)
// ---------------------------------------------------------------------------
__global__ void mean_kernel(const float* __restrict__ sum, float* __restrict__ out)
{
    int b = blockIdx.x;
    for (int j = threadIdx.x; j < DF; j += blockDim.x) {
        float acc = 0.f;
        for (int t = 0; t < TT; ++t)
            acc += sum[((size_t)b * TT + t) * DF + j];
        out[(size_t)b * DF + j] = acc * (1.0f / TT);
    }
}

// ---------------------------------------------------------------------------
// Host orchestration. Input order (setup_inputs dict order, flattened):
// x, context, poi_emb, time_emb, space_W, space_b, then (forw, back) x 5
// layers x 19 arrays: ln1_g, ln1_b, in_W, in_b, conv_W, conv_b, x_W, dt_W,
// dt_b, A_log, D, out_W, out_b, ln2_g, ln2_b, mlp_W1, mlp_b1, mlp_W2, mlp_b2
// ---------------------------------------------------------------------------
extern "C" void kernel_launch(void* const* d_in, const int* in_sizes, int n_in,
                              void* d_out, int out_size, void* d_ws, size_t ws_size,
                              hipStream_t stream)
{
    (void)in_sizes; (void)n_in; (void)out_size; (void)ws_size;
    const int*   x   = (const int*)d_in[0];
    const int*   ctx = (const int*)d_in[1];
    const float* poi = (const float*)d_in[2];
    const float* tim = (const float*)d_in[3];
    const float* sW  = (const float*)d_in[4];
    const float* sb  = (const float*)d_in[5];

    float* ws = (float*)d_ws;
    size_t off = 0;
    float* resid = ws + off; off += (size_t)MROWS * DF;
    float* lnbuf = ws + off; off += (size_t)MROWS * DF;
    float* xz    = ws + off; off += (size_t)MROWS * 2 * DI;   // also MLP hidden
    float* xh    = ws + off; off += (size_t)MROWS * DI;
    float* xdbl  = ws + off; off += (size_t)MROWS * XDBL_LD;
    float* dtb   = ws + off; off += (size_t)MROWS * DI;
    float* ybuf  = ws + off; off += (size_t)MROWS * DI;
    float* sumb  = ws + off; off += (size_t)MROWS * DF;
    _Float16* xWp  = (_Float16*)(ws + off); off += (size_t)DI * 64 / 2 + 16;
    _Float16* dtWp = (_Float16*)(ws + off); off += (size_t)32 * DI / 2 + 16;
    _Float16* whb  = (_Float16*)(ws + off); off += (size_t)(DF * 2 * DI) / 2 + 16; // fp16 weight scratch

    for (int dir = 0; dir < 2; ++dir) {
        embed_kernel<<<(MROWS * DF + 255) / 256, 256, 0, stream>>>(
            resid, x, ctx, poi, tim, sW, sb, dir);

        for (int L = 0; L < NDEPTH; ++L) {
            int base = 6 + (dir * NDEPTH + L) * 19;
            const float* ln1_g  = (const float*)d_in[base + 0];
            const float* ln1_b  = (const float*)d_in[base + 1];
            const float* in_W   = (const float*)d_in[base + 2];
            const float* in_b   = (const float*)d_in[base + 3];
            const float* conv_W = (const float*)d_in[base + 4];
            const float* conv_b = (const float*)d_in[base + 5];
            const float* x_W    = (const float*)d_in[base + 6];
            const float* dt_W   = (const float*)d_in[base + 7];
            const float* dt_b   = (const float*)d_in[base + 8];
            const float* A_log  = (const float*)d_in[base + 9];
            const float* Dp     = (const float*)d_in[base + 10];
            const float* out_W  = (const float*)d_in[base + 11];
            const float* out_b  = (const float*)d_in[base + 12];
            const float* ln2_g  = (const float*)d_in[base + 13];
            const float* ln2_b  = (const float*)d_in[base + 14];
            const float* W1     = (const float*)d_in[base + 15];
            const float* b1     = (const float*)d_in[base + 16];
            const float* W2     = (const float*)d_in[base + 17];
            const float* b2     = (const float*)d_in[base + 18];

            pack_w_kernel<<<(DI * 64 + 32 * DI + 255) / 256, 256, 0, stream>>>(
                x_W, dt_W, xWp, dtWp);

            // --- Mamba branch ---
            layernorm_kernel<<<MROWS, 128, 0, stream>>>(resid, lnbuf, ln1_g, ln1_b);

            // xz = ln @ in_W + in_b                       (8192x352 @ 352x1408)
            cvt_w_kernel<<<(DF * 2 * DI + 255) / 256, 256, 0, stream>>>(in_W, whb, DF * 2 * DI);
            gemm_wmma<<<dim3(2 * DI / 32, MROWS / 128), 128, 0, stream>>>(
                lnbuf, DF, whb, 2 * DI, in_b, nullptr, xz, 2 * DI, DF, 0);

            conv_silu_kernel<<<(MROWS * DI + 255) / 256, 256, 0, stream>>>(
                xz, conv_W, conv_b, xh);

            // xdbl = xh @ xW_pad                          (8192x704 @ 704x64)
            gemm_wmma<<<dim3(64 / 32, MROWS / 128), 128, 0, stream>>>(
                xh, DI, xWp, 64, nullptr, nullptr, xdbl, 64, DI, 0);

            // dt = softplus(xdbl[:, :32] @ dtW_pad + dt_b) (8192x32 @ 32x704)
            gemm_wmma<<<dim3(DI / 32, MROWS / 128), 128, 0, stream>>>(
                xdbl, XDBL_LD, dtWp, DI, dt_b, nullptr, dtb, DI, 32, 2);

            scan_kernel<<<(BB * DI + 255) / 256, 256, 0, stream>>>(
                dtb, xh, xdbl, xz, A_log, Dp, ybuf);

            // resid = y @ out_W + out_b + resid           (8192x704 @ 704x352)
            cvt_w_kernel<<<(DI * DF + 255) / 256, 256, 0, stream>>>(out_W, whb, DI * DF);
            gemm_wmma<<<dim3(DF / 32, MROWS / 128), 128, 0, stream>>>(
                ybuf, DI, whb, DF, out_b, resid, resid, DF, DI, 0);

            // --- MLP branch ---
            layernorm_kernel<<<MROWS, 128, 0, stream>>>(resid, lnbuf, ln2_g, ln2_b);

            // hidden = gelu(ln @ W1 + b1)                 (8192x352 @ 352x704)
            cvt_w_kernel<<<(DF * 2 * DF + 255) / 256, 256, 0, stream>>>(W1, whb, DF * 2 * DF);
            gemm_wmma<<<dim3(2 * DF / 32, MROWS / 128), 128, 0, stream>>>(
                lnbuf, DF, whb, 2 * DF, b1, nullptr, xz, 2 * DF, DF, 1);

            // resid = hidden @ W2 + b2 + resid            (8192x704 @ 704x352)
            cvt_w_kernel<<<(2 * DF * DF + 255) / 256, 256, 0, stream>>>(W2, whb, 2 * DF * DF);
            gemm_wmma<<<dim3(DF / 32, MROWS / 128), 128, 0, stream>>>(
                xz, 2 * DF, whb, DF, b2, resid, resid, DF, 2 * DF, 0);
        }
        accum_kernel<<<(MROWS * DF + 255) / 256, 256, 0, stream>>>(resid, sumb, dir);
    }
    mean_kernel<<<BB, 352, 0, stream>>>(sumb, (float*)d_out);
}
